// IEEEFraudHeteroGNN_23295902613611
// MI455X (gfx1250) — matrix operand order repaired and checked by reading, not compiled, not run on hardware
//
#include <hip/hip_runtime.h>
#include <stdint.h>

// ---------------- types ----------------
typedef __attribute__((ext_vector_type(16))) __bf16 v16bf;
typedef __attribute__((ext_vector_type(8)))  float  v8f;
typedef __attribute__((ext_vector_type(4)))  unsigned int v4u;
typedef __attribute__((ext_vector_type(8)))  unsigned int v8u;

#define HDIM 128
#define E_EDGES 400000
#define MT 200000
#define MC 20000
#define MA_REAL 5000
#define MA 5008          // padded to multiple of 16
#define ME 10000
#define TXN_IN 256

__device__ __forceinline__ unsigned short f2bf(float f) {
  unsigned int u = __builtin_bit_cast(unsigned int, f);
  unsigned int r = u + 0x7FFFu + ((u >> 16) & 1u);   // round-to-nearest-even
  return (unsigned short)(r >> 16);
}

// ---------------- elementwise helpers ----------------
__global__ void k_zero_f32(float* __restrict__ p, long n) {
  long t = (long)blockIdx.x * blockDim.x + threadIdx.x;
  if (t < n) p[t] = 0.0f;
}

__global__ void k_f32_to_bf16(const float* __restrict__ s, unsigned short* __restrict__ d, long n) {
  long t = (long)blockIdx.x * blockDim.x + threadIdx.x;
  if (t < n) d[t] = f2bf(s[t]);
}

// copy rows into padded f32 buffer + bf16 mirror (pad rows -> 0)
__global__ void k_copy_pad(const float* __restrict__ src, float* __restrict__ xf,
                           unsigned short* __restrict__ xb, int rows, int rows_pad) {
  long t = (long)blockIdx.x * blockDim.x + threadIdx.x;
  long n = (long)rows_pad * HDIM;
  if (t >= n) return;
  int r = (int)(t >> 7);
  float v = (r < rows) ? src[t] : 0.0f;
  xf[t] = v;
  xb[t] = f2bf(v);
}

// batched transpose+convert: W [K,N] f32 (row-major) -> Wt [N,K] bf16
__global__ void k_transpose_bf16(const float* __restrict__ W, unsigned short* __restrict__ Wt,
                                 int K, int N, int nmats) {
  long per = (long)K * N;
  long n = per * nmats;
  long t = (long)blockIdx.x * blockDim.x + threadIdx.x;
  if (t >= n) return;
  int m = (int)(t / per);
  long r = t % per;
  int nn = (int)(r / K);
  int kk = (int)(r % K);
  Wt[t] = f2bf(W[(long)m * per + (long)kk * N + nn]);
}

// combine conv_Wr per destination type, transpose, convert: out [2][4][N=128][K=128]
__global__ void k_wr_comb(const float* __restrict__ Wr, unsigned short* __restrict__ Wt) {
  const long per = 128 * 128;
  long t = (long)blockIdx.x * blockDim.x + threadIdx.x;
  if (t >= 2 * 4 * per) return;
  int l = (int)(t / (4 * per));
  long r = t % (4 * per);
  int d = (int)(r / per);
  long e = r % per;
  int n = (int)(e / 128), k = (int)(e % 128);
  long base = (long)l * 6 * per;
  long idx = (long)k * 128 + n;
  float s;
  if (d == 0) s = Wr[base + 1 * per + idx] + Wr[base + 3 * per + idx] + Wr[base + 5 * per + idx];
  else { int i = (d == 1) ? 0 : ((d == 2) ? 2 : 4); s = Wr[base + i * per + idx]; }
  Wt[t] = f2bf(s);
}

// combined lin_l bias per destination type: out [2][4][128]
__global__ void k_bias_comb(const float* __restrict__ bl, float* __restrict__ bc) {
  int t = blockIdx.x * blockDim.x + threadIdx.x;
  if (t >= 2 * 4 * 128) return;
  int l = t / 512, r = t % 512, d = r / 128, h = r % 128;
  int base = l * 6 * 128;
  float s;
  if (d == 0) s = bl[base + 1 * 128 + h] + bl[base + 3 * 128 + h] + bl[base + 5 * 128 + h];
  else { int i = (d == 1) ? 0 : ((d == 2) ? 2 : 4); s = bl[base + i * 128 + h]; }
  bc[t] = s;
}

// ---------------- segment mean ----------------
__global__ void k_scatter(const float* __restrict__ xs, const int* __restrict__ si,
                          const int* __restrict__ di, float* __restrict__ msum,
                          float* __restrict__ mcnt) {
  long t = (long)blockIdx.x * blockDim.x + threadIdx.x;
  int e = (int)(t >> 5);
  int c = (int)(t & 31);
  if (e >= E_EDGES) return;
  int s = si[e], d = di[e];
  const float4 v = *(const float4*)(xs + (long)s * HDIM + c * 4);
  float* p = msum + (long)d * HDIM + c * 4;
  __hip_atomic_fetch_add(p + 0, v.x, __ATOMIC_RELAXED, __HIP_MEMORY_SCOPE_AGENT);
  __hip_atomic_fetch_add(p + 1, v.y, __ATOMIC_RELAXED, __HIP_MEMORY_SCOPE_AGENT);
  __hip_atomic_fetch_add(p + 2, v.z, __ATOMIC_RELAXED, __HIP_MEMORY_SCOPE_AGENT);
  __hip_atomic_fetch_add(p + 3, v.w, __ATOMIC_RELAXED, __HIP_MEMORY_SCOPE_AGENT);
  if (c == 0)
    __hip_atomic_fetch_add(mcnt + d, 1.0f, __ATOMIC_RELAXED, __HIP_MEMORY_SCOPE_AGENT);
}

__global__ void k_finalize(const float* __restrict__ msum, const float* __restrict__ mcnt,
                           unsigned short* __restrict__ mb, int rows_pad) {
  long t = (long)blockIdx.x * blockDim.x + threadIdx.x;
  long n = (long)rows_pad * HDIM;
  if (t >= n) return;
  int r = (int)(t >> 7);
  float c = mcnt[r];
  mb[t] = f2bf(msum[t] / fmaxf(c, 1.0f));
}

// ---------------- WMMA GEMM with TDM-staged A tile ----------------
// C[M,128] (f32) {=, +=} A[M,K](bf16) @ Wt^T where Wt is [128,K] bf16 (pre-transposed B)
// flags: bit0 = accumulate into C, bit1 = add bias + relu.  Cb optional bf16 mirror.
// The 16xK bf16 A tile of this block is DMA'd once into LDS by the Tensor Data
// Mover (2-D D# descriptor, 2-group form), tracked via TENSORcnt; waves then
// read fragments via ds_load_b128 while B streams from L2-resident weights.
__global__ __launch_bounds__(256) void k_gemm_wmma(
    const unsigned short* __restrict__ A, const unsigned short* __restrict__ Wt,
    float* __restrict__ C, unsigned short* __restrict__ Cb,
    const float* __restrict__ bias, int K, int flags) {
  const int N = HDIM;
  __shared__ __align__(16) unsigned short ldsA[16 * TXN_IN];  // up to 16x256 bf16 = 8 KB

  int wave = threadIdx.x >> 5;
  int lane = threadIdx.x & 31;
  int g = lane >> 4;          // half-wave group
  int ln15 = lane & 15;
  int row0 = blockIdx.x * 16;
  int n0 = wave * 16;         // 8 waves cover N=128

  // ---- TDM: issue one tensor_load_to_lds for the 16xK A tile (wave 0 only) ----
  if (threadIdx.x < 32) {
    unsigned long long gaddr = (unsigned long long)(uintptr_t)(A + (size_t)row0 * K);
    unsigned lds_off = (unsigned)(uintptr_t)(void*)&ldsA[0];  // flat LDS addr low 32b = LDS offset
    unsigned td0 = (unsigned)K;     // tensor_dim0 (elements)
    unsigned td1 = 16u;             // tensor_dim1 (rows)
    v4u g0;
    g0[0] = 1u;                                         // count=1, user descriptor
    g0[1] = lds_off;                                    // lds_addr
    g0[2] = (unsigned)gaddr;                            // global_addr[31:0]
    g0[3] = (unsigned)((gaddr >> 32) & 0x01FFFFFFull)   // global_addr[56:32]
            | 0x80000000u;                              // type=2 ("image")
    v8u g1;
    g1[0] = 0x00010000u;                  // workgroup_mask=0, data_size=1 (2-byte elems)
    g1[1] = (td0 & 0xFFFFu) << 16;        // tensor_dim0[15:0]
    g1[2] = (td0 >> 16) | ((td1 & 0xFFFFu) << 16);   // tensor_dim0[31:16] | tensor_dim1[15:0]
    g1[3] = (td1 >> 16) | ((unsigned)K << 16);       // tensor_dim1[31:16] | tile_dim0
    g1[4] = 16u;                          // tile_dim1=16, tile_dim2=0
    g1[5] = (unsigned)K;                  // tensor_dim0_stride[31:0]
    g1[6] = 0u;                           // stride0[47:32] | stride1[15:0]
    g1[7] = 0u;                           // stride1[47:16]
    asm volatile("tensor_load_to_lds %0, %1" :: "s"(g0), "s"(g1) : "memory");
    __builtin_amdgcn_s_wait_tensorcnt(0);
  }
  __syncthreads();

  const unsigned short* arow = ldsA + (long)ln15 * K;          // LDS (ds_load path)
  const unsigned short* brow = Wt + (long)(n0 + ln15) * K;     // global (L2-resident)

  union Frag { v16bf v; uint4 q[2]; };
  Frag a, b;

  float* crow = C + (long)(row0 + g * 8) * N + n0 + ln15;
  v8f c;
  if (flags & 1) {
#pragma unroll
    for (int r = 0; r < 8; ++r) c[r] = crow[(long)r * N];
  } else {
#pragma unroll
    for (int r = 0; r < 8; ++r) c[r] = 0.0f;
  }

  for (int kt = 0; kt < K; kt += 32) {
    // A 16x32 bf16 fragment: lane(g=0) rows hold K = kt+[0..7], kt+[16..23]; g=1: +8
    a.q[0] = *(const uint4*)(arow + kt + 8 * g);
    a.q[1] = *(const uint4*)(arow + kt + 16 + 8 * g);
    // B 32x16 bf16 fragment from [N,K]-major weights: K = kt + 8*g + [0..7], kt + 16 + 8*g + [0..7]
    b.q[0] = *(const uint4*)(brow + kt + 8 * g);
    b.q[1] = *(const uint4*)(brow + kt + 16 + 8 * g);
    c = __builtin_amdgcn_wmma_f32_16x16x32_bf16(false, a.v, false, b.v,
                                                (short)0, c, false, false);
  }

  if (flags & 2) {
    float bv = bias[n0 + ln15];
#pragma unroll
    for (int r = 0; r < 8; ++r) {
      float v = c[r] + bv;
      c[r] = v > 0.0f ? v : 0.0f;
    }
  }
#pragma unroll
  for (int r = 0; r < 8; ++r) crow[(long)r * N] = c[r];
  if (Cb) {
    unsigned short* cb = Cb + (long)(row0 + g * 8) * N + n0 + ln15;
#pragma unroll
    for (int r = 0; r < 8; ++r) cb[(long)r * N] = f2bf(c[r]);
  }
}

// ---------------- LayerNorm + ReLU (wave per row) ----------------
__global__ void k_ln_relu(const float* __restrict__ x, const float* __restrict__ bc,
                          const float* __restrict__ g, const float* __restrict__ b,
                          float* __restrict__ xf, unsigned short* __restrict__ xb, int rows) {
  int wv = (int)(((long)blockIdx.x * blockDim.x + threadIdx.x) >> 5);
  int lane = threadIdx.x & 31;
  if (wv >= rows) return;
  float4 v = *(const float4*)(x + (long)wv * HDIM + lane * 4);
  const float4 a = *(const float4*)(bc + lane * 4);
  v.x += a.x; v.y += a.y; v.z += a.z; v.w += a.w;
  float s = v.x + v.y + v.z + v.w;
  for (int o = 16; o; o >>= 1) s += __shfl_xor(s, o);
  float mu = s * (1.0f / 128.0f);
  float d0 = v.x - mu, d1 = v.y - mu, d2 = v.z - mu, d3 = v.w - mu;
  float q = d0 * d0 + d1 * d1 + d2 * d2 + d3 * d3;
  for (int o = 16; o; o >>= 1) q += __shfl_xor(q, o);
  float inv = rsqrtf(q * (1.0f / 128.0f) + 1e-5f);
  const float4 gg = *(const float4*)(g + lane * 4);
  const float4 bb = *(const float4*)(b + lane * 4);
  float y0 = fmaxf(d0 * inv * gg.x + bb.x, 0.0f);
  float y1 = fmaxf(d1 * inv * gg.y + bb.y, 0.0f);
  float y2 = fmaxf(d2 * inv * gg.z + bb.z, 0.0f);
  float y3 = fmaxf(d3 * inv * gg.w + bb.w, 0.0f);
  float4 out; out.x = y0; out.y = y1; out.z = y2; out.w = y3;
  *(float4*)(xf + (long)wv * HDIM + lane * 4) = out;
  uint2 pk;
  pk.x = (unsigned)f2bf(y0) | ((unsigned)f2bf(y1) << 16);
  pk.y = (unsigned)f2bf(y2) | ((unsigned)f2bf(y3) << 16);
  *(uint2*)(xb + (long)wv * HDIM + lane * 4) = pk;
}

// ---------------- head (wave per row) ----------------
__global__ void k_head(const float* __restrict__ xf, const float* __restrict__ Wh,
                       const float* __restrict__ bh, float* __restrict__ out, int rows) {
  int wv = (int)(((long)blockIdx.x * blockDim.x + threadIdx.x) >> 5);
  int lane = threadIdx.x & 31;
  if (wv >= rows) return;
  const float4 v = *(const float4*)(xf + (long)wv * HDIM + lane * 4);
  const float4 w = *(const float4*)(Wh + lane * 4);
  float s = v.x * w.x + v.y * w.y + v.z * w.z + v.w * w.w;
  for (int o = 16; o; o >>= 1) s += __shfl_xor(s, o);
  if (lane == 0) out[wv] = s + bh[0];
}

// ---------------- host ----------------
extern "C" void kernel_launch(void* const* d_in, const int* in_sizes, int n_in,
                              void* d_out, int out_size, void* d_ws, size_t ws_size,
                              hipStream_t stream) {
  const float* txn_x    = (const float*)d_in[0];
  const float* emb_card = (const float*)d_in[1];
  const float* emb_addr = (const float*)d_in[2];
  const float* emb_email= (const float*)d_in[3];
  const float* W_enc    = (const float*)d_in[4];
  const float* b_enc    = (const float*)d_in[5];
  const float* conv_Wl  = (const float*)d_in[6];
  const float* conv_bl  = (const float*)d_in[7];
  const float* conv_Wr  = (const float*)d_in[8];
  const float* ln_g     = (const float*)d_in[9];
  const float* ln_b     = (const float*)d_in[10];
  const float* W_head   = (const float*)d_in[11];
  const float* b_head   = (const float*)d_in[12];

  // relations: i -> (src node type, dst node type, src idx, dst idx)
  const int rel_src[6] = {0, 1, 0, 2, 0, 3};
  const int rel_dst[6] = {1, 0, 2, 0, 3, 0};
  const int* rel_si[6] = {(const int*)d_in[13], (const int*)d_in[14], (const int*)d_in[15],
                          (const int*)d_in[16], (const int*)d_in[17], (const int*)d_in[18]};
  const int* rel_di[6] = {(const int*)d_in[14], (const int*)d_in[13], (const int*)d_in[16],
                          (const int*)d_in[15], (const int*)d_in[18], (const int*)d_in[17]};

  const int rows_pad[4] = {MT, MC, MA, ME};

  // ---- workspace carve ----
  char* base = (char*)d_ws;
  size_t off = 0;
  auto carve = [&](size_t bytes) {
    char* p = base + off;
    off = (off + bytes + 255) & ~(size_t)255;
    return p;
  };
  float*          xf[4];
  unsigned short* xb[4];
  float*          nf[4];
  for (int d = 0; d < 4; ++d) xf[d] = (float*)carve((size_t)rows_pad[d] * HDIM * 4);
  for (int d = 0; d < 4; ++d) xb[d] = (unsigned short*)carve((size_t)rows_pad[d] * HDIM * 2);
  for (int d = 0; d < 4; ++d) nf[d] = (float*)carve((size_t)rows_pad[d] * HDIM * 4);
  float* msum = (float*)carve((size_t)MT * HDIM * 4);   // also reused for bf16 txn_x
  float* mcnt = (float*)carve((size_t)MT * 4);
  unsigned short* m_b    = (unsigned short*)carve((size_t)MT * HDIM * 2);
  unsigned short* Wenc_t = (unsigned short*)carve((size_t)TXN_IN * HDIM * 2);
  unsigned short* Wl_t   = (unsigned short*)carve((size_t)12 * HDIM * HDIM * 2);
  unsigned short* Wrc_t  = (unsigned short*)carve((size_t)8 * HDIM * HDIM * 2);
  float*          bias_c = (float*)carve((size_t)8 * HDIM * 4);

  auto nb = [](long n, int bs) { return (unsigned)((n + bs - 1) / bs); };

  // ---- weight prep (bf16, transposed / combined) ----
  k_transpose_bf16<<<nb((long)TXN_IN * HDIM, 256), 256, 0, stream>>>(W_enc, Wenc_t, TXN_IN, HDIM, 1);
  k_transpose_bf16<<<nb((long)12 * HDIM * HDIM, 256), 256, 0, stream>>>(conv_Wl, Wl_t, HDIM, HDIM, 12);
  k_wr_comb<<<nb((long)8 * HDIM * HDIM, 256), 256, 0, stream>>>(conv_Wr, Wrc_t);
  k_bias_comb<<<nb(8 * HDIM, 256), 256, 0, stream>>>(conv_bl, bias_c);

  // ---- node feature init ----
  k_copy_pad<<<nb((long)MC * HDIM, 256), 256, 0, stream>>>(emb_card, xf[1], xb[1], MC, MC);
  k_copy_pad<<<nb((long)MA * HDIM, 256), 256, 0, stream>>>(emb_addr, xf[2], xb[2], MA_REAL, MA);
  k_copy_pad<<<nb((long)ME * HDIM, 256), 256, 0, stream>>>(emb_email, xf[3], xb[3], ME, ME);

  // txn encoder: bf16-convert inputs into msum scratch, WMMA GEMM with bias+relu
  unsigned short* txnx_b = (unsigned short*)msum;
  k_f32_to_bf16<<<nb((long)MT * TXN_IN, 256), 256, 0, stream>>>(txn_x, txnx_b, (long)MT * TXN_IN);
  k_gemm_wmma<<<MT / 16, 256, 0, stream>>>(txnx_b, Wenc_t, xf[0], xb[0], b_enc, TXN_IN, /*flags=*/2);

  // ---- two hetero-SAGE layers ----
  for (int l = 0; l < 2; ++l) {
    for (int d = 0; d < 4; ++d)
      k_zero_f32<<<nb((long)rows_pad[d] * HDIM, 256), 256, 0, stream>>>(nf[d], (long)rows_pad[d] * HDIM);

    for (int i = 0; i < 6; ++i) {
      int d = rel_dst[i], s = rel_src[i];
      long nelem = (long)rows_pad[d] * HDIM;
      k_zero_f32<<<nb(nelem, 256), 256, 0, stream>>>(msum, nelem);
      k_zero_f32<<<nb(rows_pad[d], 256), 256, 0, stream>>>(mcnt, rows_pad[d]);
      k_scatter<<<nb((long)E_EDGES * 32, 256), 256, 0, stream>>>(xf[s], rel_si[i], rel_di[i], msum, mcnt);
      k_finalize<<<nb(nelem, 256), 256, 0, stream>>>(msum, mcnt, m_b, rows_pad[d]);
      k_gemm_wmma<<<rows_pad[d] / 16, 256, 0, stream>>>(
          m_b, Wl_t + (size_t)(l * 6 + i) * HDIM * HDIM, nf[d], nullptr, nullptr, HDIM, /*flags=*/1);
    }
    for (int d = 0; d < 4; ++d)
      k_gemm_wmma<<<rows_pad[d] / 16, 256, 0, stream>>>(
          xb[d], Wrc_t + (size_t)(l * 4 + d) * HDIM * HDIM, nf[d], nullptr, nullptr, HDIM, /*flags=*/1);

    for (int d = 0; d < 4; ++d)
      k_ln_relu<<<nb((long)rows_pad[d] * 32, 256), 256, 0, stream>>>(
          nf[d], bias_c + (size_t)(l * 4 + d) * HDIM, ln_g + (size_t)d * HDIM,
          ln_b + (size_t)d * HDIM, xf[d], xb[d], rows_pad[d]);
  }

  // ---- head ----
  k_head<<<nb((long)MT * 32, 256), 256, 0, stream>>>(xf[0], W_head, b_head, (float*)d_out, MT);
  (void)in_sizes; (void)n_in; (void)out_size; (void)ws_size;
}
